// SeqAttwithPairBias_34660386079027
// MI455X (gfx1250) — matrix operand order repaired and checked by reading, not compiled, not run on hardware
//
#include <hip/hip_runtime.h>

// ---------------------------------------------------------------------------
// SeqAtt-with-pair-bias for MI455X (gfx1250, wave32, WMMA).
// Roofline: pair tensor 512MB (read ONCE, non-temporal) + current_kv 134MB
// (written once, non-temporal) dominate; GEMMs are bf16 WMMA
// (v_wmma_f32_16x16x32_bf16), one wave per 16x16 C tile, f32 accumulate.
// ---------------------------------------------------------------------------

typedef __attribute__((ext_vector_type(16))) __bf16 v16bf;
typedef __attribute__((ext_vector_type(8)))  __bf16 v8bf;
typedef __attribute__((ext_vector_type(8)))  float  v8f;
typedef __attribute__((ext_vector_type(4)))  float  v4f;

#define L_   1024
#define SD_  512
#define PD_  128
#define NH_  8
#define HD_  64   // qd = kd = vd

static __device__ __forceinline__ float sigf(float x) {
  return 1.0f / (1.0f + __expf(-x));
}

// --------------------------- f32 -> bf16 copy ------------------------------
__global__ void cvt_bf16_kernel(const float* __restrict__ in,
                                __bf16* __restrict__ out, int n) {
  int i = blockIdx.x * 256 + threadIdx.x;
  if (i < n) out[i] = (__bf16)in[i];
}

// --------------------------- zero accumulators -----------------------------
__global__ void zero_kernel(float* __restrict__ p, int n) {
  int i = blockIdx.x * 256 + threadIdx.x;
  if (i < n) p[i] = 0.f;
}

// ---------------- single-pass pair reduction (row + col means) -------------
// One streaming pass over pair[1024,1024,128] (512MB, non-temporal loads).
// rowAcc[i][c] += sum_j pair[i][j][c]      (-> pair_j)
// colAcc[j][c] += sum_i pair[i][j][c]      (-> pair_i)
// Tiles: 16 rows(i) x 64 cols(j) per block; accumulators (1MB) stay in L2.
__global__ void pair_pass_kernel(const float* __restrict__ pair,
                                 float* __restrict__ rowAcc,
                                 float* __restrict__ colAcc) {
  const int j0 = blockIdx.x * 64;     // 16 j-tiles
  const int i0 = blockIdx.y * 16;     // 64 i-tiles
  const int t  = threadIdx.x;         // 256 threads
  const int c  = t & (PD_ - 1);       // channel 0..127
  const int jj = t >> 7;              // 0 or 1 (two j-rows per sweep)

  float col[32];
#pragma unroll
  for (int s = 0; s < 32; ++s) col[s] = 0.f;

  for (int i = 0; i < 16; ++i) {
    const float* base = pair + ((size_t)(i0 + i) * L_ + j0 + jj) * PD_ + c;
    float rowp = 0.f;
#pragma unroll
    for (int s = 0; s < 32; ++s) {          // j = j0 + jj + 2*s
      float v = __builtin_nontemporal_load(base + (size_t)(2 * s) * PD_);
      rowp   += v;
      col[s] += v;
    }
    atomicAdd(rowAcc + (i0 + i) * PD_ + c, rowp);
  }
#pragma unroll
  for (int s = 0; s < 32; ++s)
    atomicAdd(colAcc + (j0 + jj + 2 * s) * PD_ + c, col[s]);
}

// scale by 1/L, emit bf16 operands for the pair-bias WMMA GEMMs
__global__ void pair_finalize_kernel(const float* __restrict__ rowAcc,
                                     const float* __restrict__ colAcc,
                                     __bf16* __restrict__ pairJh,
                                     __bf16* __restrict__ pairIh) {
  int idx = blockIdx.x * 256 + threadIdx.x;
  if (idx < L_ * PD_) {
    pairJh[idx] = (__bf16)(rowAcc[idx] * (1.0f / (float)L_));
    pairIh[idx] = (__bf16)(colAcc[idx] * (1.0f / (float)L_));
  }
}

// ----------------------- LayerNorm(seq) -> bf16 ----------------------------
__global__ void ln_kernel(const float* __restrict__ seq,
                          const float* __restrict__ g,
                          const float* __restrict__ b,
                          __bf16* __restrict__ xh) {
  __shared__ float ssum[256], ssq[256];
  __shared__ float mu_s, rs_s;
  int l = blockIdx.x, t = threadIdx.x;       // 256 threads, 512 elems
  const float* row = seq + (size_t)l * SD_;
  float v0 = row[t], v1 = row[t + 256];
  ssum[t] = v0 + v1;
  ssq[t]  = v0 * v0 + v1 * v1;
  __syncthreads();
  for (int s = 128; s > 0; s >>= 1) {
    if (t < s) { ssum[t] += ssum[t + s]; ssq[t] += ssq[t + s]; }
    __syncthreads();
  }
  if (t == 0) {
    float mu  = ssum[0] * (1.0f / SD_);
    float var = ssq[0] * (1.0f / SD_) - mu * mu;  // biased, matches torch LN
    mu_s = mu;
    rs_s = rsqrtf(var + 1e-5f);
  }
  __syncthreads();
  float mu = mu_s, rs = rs_s;
  xh[(size_t)l * SD_ + t]       = (__bf16)((v0 - mu) * rs * g[t]       + b[t]);
  xh[(size_t)l * SD_ + t + 256] = (__bf16)((v1 - mu) * rs * g[t + 256] + b[t + 256]);
}

// --------------------------- WMMA bf16 GEMM --------------------------------
// C[M,N] = A[M,K](bf16,row-major) * B^T where B=W[N,K](bf16,row-major),
// + bias[N], optional sigmoid. One wave per 16x16 tile.
__global__ void __launch_bounds__(32)
wmma_gemm_kernel(const __bf16* __restrict__ A, const __bf16* __restrict__ W,
                 const float* __restrict__ bias, float* __restrict__ C,
                 int M, int N, int Kd, int act) {
  const int lane = threadIdx.x & 31;
  const int half = lane >> 4;     // 0: lanes 0-15, 1: lanes 16-31
  const int r    = lane & 15;
  const int n0   = blockIdx.x * 16;
  const int m0   = blockIdx.y * 16;

  v8f c = {};
  const __bf16* arow = A + (size_t)(m0 + r) * Kd;
  const __bf16* brow = W + (size_t)(n0 + r) * Kd;   // B column n = W row n

  for (int kk = 0; kk < Kd; kk += 32) {
    // A 16x32 bf16 fragment: lanes<16 hold K 0..15, lanes>=16 K 16..31
    v16bf a = *(const v16bf*)(arow + kk + half * 16);
    // B 32x16 bf16 fragment: lane = column N; lanes<16 hold K 0..15, lanes>=16 K 16..31
    v16bf bfrag = *(const v16bf*)(brow + kk + half * 16);
    c = __builtin_amdgcn_wmma_f32_16x16x32_bf16(
        /*neg_a=*/false, a, /*neg_b=*/false, bfrag,
        /*c_mod=*/(short)0, c, /*reuse_a=*/false, /*reuse_b=*/false);
  }

#pragma unroll
  for (int e = 0; e < 8; ++e) {
    int row = m0 + (half << 3) + e;     // C/D layout: VGPR e -> M = e + 8*half
    int col = n0 + r;
    float v = c[e] + bias[col];
    if (act) v = sigf(v);
    C[(size_t)row * N + col] = v;
  }
}

// --------- current_kv outer products + factored s = Q . kv -----------------
// kv[n,l,k,v] = decay_n * pi_k * pj_v + K_k * V_v     (non-temporal stores)
// s[n,l,v]    = decay_n * (Q.pi) * pj_v + (Q.K) * V_v (no kv re-read needed)
__global__ void kv_s_kernel(const float* __restrict__ Q,
                            const float* __restrict__ Kf,
                            const float* __restrict__ Vf,
                            const float* __restrict__ PI,
                            const float* __restrict__ PJ,
                            float* __restrict__ kv_out,
                            float* __restrict__ S) {
  int nl = blockIdx.x;                 // n*1024 + l
  int n = nl >> 10, l = nl & (L_ - 1);
  __shared__ float q[HD_], k[HD_], v[HD_], pi[HD_], pj[HD_];
  __shared__ float rqk[HD_], rqp[HD_];
  int t = threadIdx.x;                 // 256 threads
  size_t base = (size_t)l * (NH_ * HD_) + n * HD_;
  if (t < HD_) {
    q[t]  = Q[base + t];  k[t]  = Kf[base + t]; v[t] = Vf[base + t];
    pi[t] = PI[base + t]; pj[t] = PJ[base + t];
    rqk[t] = Q[base + t] * Kf[base + t];
    rqp[t] = Q[base + t] * PI[base + t];
  }
  __syncthreads();
  for (int s = 32; s > 0; s >>= 1) {   // tree-reduce the two dot products
    if (t < s) { rqk[t] += rqk[t + s]; rqp[t] += rqp[t + s]; }
    __syncthreads();
  }
  float dqk = rqk[0], dqp = rqp[0];
  float decay = 1.0f - exp2f(-5.0f - (float)n);
  // 4096 kv elems / 256 threads = 16 each; constant k per thread, v4 NT stores
  int kc = t >> 2;
  int v0 = (t & 3) * 16;
  float pik = decay * pi[kc];
  float kkc = k[kc];
  v4f* dst = (v4f*)(kv_out + (size_t)nl * (HD_ * HD_) + kc * HD_ + v0);
#pragma unroll
  for (int e4 = 0; e4 < 4; ++e4) {
    v4f o;
#pragma unroll
    for (int e = 0; e < 4; ++e)
      o[e] = pik * pj[v0 + e4 * 4 + e] + kkc * v[v0 + e4 * 4 + e];
    __builtin_nontemporal_store(o, dst + e4);   // 134MB streamed once
  }
  if (t < HD_)
    S[(size_t)nl * HD_ + t] = decay * dqp * pj[t] + dqk * v[t];
}

// ------------------- per-head group-norm statistics ------------------------
__global__ void head_stats_kernel(const float* __restrict__ S,
                                  float* __restrict__ stats) {
  __shared__ float ssum[256], ssq[256];
  int n = blockIdx.x, t = threadIdx.x;
  const float* p = S + (size_t)n * (L_ * HD_);
  float a = 0.f, b = 0.f;
  for (int i = t; i < L_ * HD_; i += 256) { float x = p[i]; a += x; b += x * x; }
  ssum[t] = a; ssq[t] = b;
  __syncthreads();
  for (int s = 128; s > 0; s >>= 1) {
    if (t < s) { ssum[t] += ssum[t + s]; ssq[t] += ssq[t + s]; }
    __syncthreads();
  }
  if (t == 0) {
    float mu  = ssum[0] * (1.0f / (L_ * HD_));
    float var = ssq[0] * (1.0f / (L_ * HD_)) - mu * mu;
    stats[2 * n]     = mu;
    stats[2 * n + 1] = rsqrtf(var + 1e-5f);
  }
}

// -------- gate * sigmoid(gate) * normalized(s) -> bf16 [l, n*64+v] ---------
__global__ void gate_y_kernel(const float* __restrict__ G,
                              const float* __restrict__ S,
                              const float* __restrict__ stats,
                              __bf16* __restrict__ yh) {
  int l = blockIdx.x, t = threadIdx.x;
  for (int o = t; o < NH_ * HD_; o += 256) {
    int n = o >> 6, vv = o & (HD_ - 1);
    float g  = G[(size_t)l * (NH_ * HD_) + o];            // already sigmoided
    float sn = (S[((size_t)n * L_ + l) * HD_ + vv] - stats[2 * n]) * stats[2 * n + 1];
    float y  = g * sigf(g) * sn;
    yh[(size_t)l * (NH_ * HD_) + o] = (__bf16)y;
  }
}

// ---------------------------------------------------------------------------
extern "C" void kernel_launch(void* const* d_in, const int* in_sizes, int n_in,
                              void* d_out, int out_size, void* d_ws, size_t ws_size,
                              hipStream_t stream) {
  const float* seq  = (const float*)d_in[0];
  const float* pair = (const float*)d_in[1];
  const float* Wq   = (const float*)d_in[2];
  const float* bq   = (const float*)d_in[3];
  const float* Wk   = (const float*)d_in[4];
  const float* bk   = (const float*)d_in[5];
  const float* Wv   = (const float*)d_in[6];
  const float* bv   = (const float*)d_in[7];
  const float* Wbi  = (const float*)d_in[8];
  const float* bbi  = (const float*)d_in[9];
  const float* Wbj  = (const float*)d_in[10];
  const float* bbj  = (const float*)d_in[11];
  const float* Wg   = (const float*)d_in[12];
  const float* bg   = (const float*)d_in[13];
  const float* Wo   = (const float*)d_in[14];
  const float* bo   = (const float*)d_in[15];
  const float* ln_g = (const float*)d_in[16];
  const float* ln_b = (const float*)d_in[17];

  float* out_seq = (float*)d_out;                         // [1024,512]
  float* out_kv  = (float*)d_out + (size_t)L_ * SD_;      // [8,1024,64,64]

  // ---- workspace carve-up ----
  char* w = (char*)d_ws;
  float*  rowAcc = (float*)w;               w += (size_t)L_ * PD_ * 4;   // 512KB
  float*  colAcc = (float*)w;               w += (size_t)L_ * PD_ * 4;   // (adjacent; zeroed together)
  __bf16* pairIh = (__bf16*)w;              w += (size_t)L_ * PD_ * 2;   // 256KB
  __bf16* pairJh = (__bf16*)w;              w += (size_t)L_ * PD_ * 2;
  __bf16* xh     = (__bf16*)w;              w += (size_t)L_ * SD_ * 2;   // 1MB
  __bf16* Wqh    = (__bf16*)w;              w += (size_t)SD_ * SD_ * 2;  // 512KB
  __bf16* Wkh    = (__bf16*)w;              w += (size_t)SD_ * SD_ * 2;
  __bf16* Wvh    = (__bf16*)w;              w += (size_t)SD_ * SD_ * 2;
  __bf16* Wgh    = (__bf16*)w;              w += (size_t)SD_ * SD_ * 2;
  __bf16* Wbih   = (__bf16*)w;              w += (size_t)SD_ * PD_ * 2;  // 128KB
  __bf16* Wbjh   = (__bf16*)w;              w += (size_t)SD_ * PD_ * 2;
  __bf16* Woh    = (__bf16*)w;              w += (size_t)SD_ * SD_ * 2;
  float*  Qf     = (float*)w;               w += (size_t)L_ * SD_ * 4;   // 2MB
  float*  Kff    = (float*)w;               w += (size_t)L_ * SD_ * 4;
  float*  Vff    = (float*)w;               w += (size_t)L_ * SD_ * 4;
  float*  Gf     = (float*)w;               w += (size_t)L_ * SD_ * 4;
  float*  PIf    = (float*)w;               w += (size_t)L_ * SD_ * 4;
  float*  PJf    = (float*)w;               w += (size_t)L_ * SD_ * 4;
  float*  Sf     = (float*)w;               w += (size_t)NH_ * L_ * HD_ * 4; // 2MB
  __bf16* yh     = (__bf16*)w;              w += (size_t)L_ * SD_ * 2;   // 1MB
  float*  stats  = (float*)w;               w += 256;

  // ---- 0) zero the pair accumulators (they persist across graph replays) ----
  zero_kernel<<<(2 * L_ * PD_ + 255) / 256, 256, 0, stream>>>(rowAcc, 2 * L_ * PD_);

  // ---- 1) weights -> bf16 ----
  auto cvt = [&](const float* src, __bf16* dst, int n) {
    cvt_bf16_kernel<<<(n + 255) / 256, 256, 0, stream>>>(src, dst, n);
  };
  cvt(Wq,  Wqh,  SD_ * SD_);
  cvt(Wk,  Wkh,  SD_ * SD_);
  cvt(Wv,  Wvh,  SD_ * SD_);
  cvt(Wg,  Wgh,  SD_ * SD_);
  cvt(Wbi, Wbih, SD_ * PD_);
  cvt(Wbj, Wbjh, SD_ * PD_);
  cvt(Wo,  Woh,  SD_ * SD_);

  // ---- 2) single 512MB streaming pass: both pair reductions ----
  {
    dim3 g(L_ / 64, L_ / 16);   // 16 j-tiles x 64 i-tiles = 1024 blocks
    pair_pass_kernel<<<g, 256, 0, stream>>>(pair, rowAcc, colAcc);
    pair_finalize_kernel<<<(L_ * PD_ + 255) / 256, 256, 0, stream>>>(
        rowAcc, colAcc, pairJh, pairIh);
  }

  // ---- 3) LayerNorm(seq) -> bf16 ----
  ln_kernel<<<L_, 256, 0, stream>>>(seq, ln_g, ln_b, xh);

  // ---- 4) WMMA GEMMs ----
  dim3 g512(SD_ / 16, L_ / 16);   // N=512, M=1024
  wmma_gemm_kernel<<<g512, 32, 0, stream>>>(xh, Wqh, bq, Qf,  L_, SD_, SD_, 0);
  wmma_gemm_kernel<<<g512, 32, 0, stream>>>(xh, Wkh, bk, Kff, L_, SD_, SD_, 0);
  wmma_gemm_kernel<<<g512, 32, 0, stream>>>(xh, Wvh, bv, Vff, L_, SD_, SD_, 0);
  wmma_gemm_kernel<<<g512, 32, 0, stream>>>(xh, Wgh, bg, Gf,  L_, SD_, SD_, 1); // gate=sigmoid
  wmma_gemm_kernel<<<g512, 32, 0, stream>>>(pairIh, Wbih, bbi, PIf, L_, SD_, PD_, 0);
  wmma_gemm_kernel<<<g512, 32, 0, stream>>>(pairJh, Wbjh, bbj, PJf, L_, SD_, PD_, 0);

  // ---- 5) current_kv outer products (NT stores) + factored s ----
  kv_s_kernel<<<NH_ * L_, 256, 0, stream>>>(Qf, Kff, Vff, PIf, PJf, out_kv, Sf);

  // ---- 6) per-head group-norm stats ----
  head_stats_kernel<<<NH_, 256, 0, stream>>>(Sf, stats);

  // ---- 7) gate * silu * norm -> bf16 ----
  gate_y_kernel<<<L_, 256, 0, stream>>>(Gf, Sf, stats, yh);

  // ---- 8) output projection (WMMA) ----
  wmma_gemm_kernel<<<g512, 32, 0, stream>>>(yh, Woh, bo, out_seq, L_, SD_, SD_, 0);
}
